// DilateResUNetCLMemMLPPH_11725260718555
// MI455X (gfx1250) — matrix operand order repaired and checked by reading, not compiled, not run on hardware
//
#include <hip/hip_runtime.h>
#include <hip/hip_bf16.h>
#include <stdint.h>

typedef __attribute__((ext_vector_type(16))) _Float16 v16h;
typedef __attribute__((ext_vector_type(8)))  float    v8f;

#define HW      16384      // 128*128 pixels
#define CCH     512        // channels
#define NCLASS  5
#define IGNORE_ 5
#define MVBG    4096
#define MV      2048
#define TOT     12288      // 4096 + 4*2048 slots
#define DOUT    256        // final feature dim

union FragU { unsigned int u[8]; v16h h; };

static __device__ __forceinline__ v8f wmma_f16(const FragU& a, const FragU& b, v8f c) {
  return __builtin_amdgcn_wmma_f32_16x16x32_f16(
      /*neg_a=*/false, a.h, /*neg_b=*/false, b.h,
      /*c_mod=*/(short)0, c, /*reuse_a=*/false, /*reuse_b=*/false);
}

// ---------------------------------------------------------------------------
// 1) strided argmax of labels[0, ::8, ::8, :]  -> y_hat (first-occurrence max)
// ---------------------------------------------------------------------------
__global__ void k_yhat(const float* __restrict__ labels, int* __restrict__ yhat) {
  int p = blockIdx.x * blockDim.x + threadIdx.x;       // 0..16383
  int i = p >> 7, j = p & 127;
  const float* L = labels + ((size_t)(i * 8) * 1024 + (size_t)(j * 8)) * NCLASS;
  float best = L[0]; int bi = 0;
  #pragma unroll
  for (int c = 1; c < NCLASS; ++c) { float v = L[c]; if (v > best) { best = v; bi = c; } }
  yhat[p] = bi;
}

// ---------------------------------------------------------------------------
// 2) stratified take-first-k selection, one block (256 thr) per class.
// ---------------------------------------------------------------------------
__global__ void k_select(const int* __restrict__ yhat, const int* __restrict__ pred,
                         int* __restrict__ sel, int* __restrict__ out_y) {
  const int c    = blockIdx.x;
  const int tid  = threadIdx.x;
  const int mv   = (c == 0) ? MVBG : MV;
  const int base = (c == 0) ? 0 : (MVBG + (c - 1) * MV);

  __shared__ int hcnt[256], ecnt[256], hoff[256], eoff[256];
  __shared__ int sh_hk, sh_ek, sh_nsel;

  int h = 0, e = 0;
  const int p0 = tid * (HW / 256);
  for (int i = 0; i < HW / 256; ++i) {
    int p = p0 + i;
    if (yhat[p] == c) { if (pred[p] == c) ++e; else ++h; }
  }
  hcnt[tid] = h; ecnt[tid] = e;
  __syncthreads();

  if (tid == 0) {
    int ah = 0, ae = 0;
    for (int t = 0; t < 256; ++t) { hoff[t] = ah; ah += hcnt[t]; eoff[t] = ae; ae += ecnt[t]; }
    int nh = ah, ne = ae;
    int cnt = nh + ne;
    int valid = (cnt > 1) ? 1 : 0;
    int nsel = ((cnt < mv) ? cnt : mv) * valid;
    bool both = (2 * nh >= nsel) && (2 * ne >= nsel);
    int hk = both ? (nsel / 2) : ((2 * nh >= nsel) ? (nsel - ne) : nh);
    sh_hk = hk; sh_ek = nsel - hk; sh_nsel = nsel;
  }
  __syncthreads();

  const int hk = sh_hk, ek = sh_ek, nsel = sh_nsel;

  int hr = hoff[tid], er = eoff[tid];
  for (int i = 0; i < HW / 256; ++i) {
    int p = p0 + i;
    if (yhat[p] == c) {
      if (pred[p] == c) { if (er < ek) sel[base + hk + er] = p; ++er; }
      else              { if (hr < hk) sel[base + hr]      = p; ++hr; }
    }
  }
  for (int s = nsel + tid; s < mv; s += 256) sel[base + s] = -1;
  for (int s = tid; s < mv; s += 256)        out_y[base + s] = (s < nsel) ? c : IGNORE_;
}

// ---------------------------------------------------------------------------
// 3) f32 -> f16 weight conversion
// ---------------------------------------------------------------------------
__global__ void k_wconv(const float* __restrict__ w1, const float* __restrict__ w2,
                        const float* __restrict__ w3,
                        _Float16* __restrict__ W1, _Float16* __restrict__ W2,
                        _Float16* __restrict__ W3) {
  int idx = blockIdx.x * 256 + threadIdx.x;            // 0..458751
  if (idx < 262144)      W1[idx]          = (_Float16)w1[idx];
  else if (idx < 393216) W2[idx - 262144] = (_Float16)w2[idx - 262144];
  else if (idx < 458752) W3[idx - 393216] = (_Float16)w3[idx - 393216];
}

// ---------------------------------------------------------------------------
// 4) gather X[slot][c] = feats[0][c][p_slot] as f16 (zero rows for padding)
// ---------------------------------------------------------------------------
__global__ void k_gather(const float* __restrict__ feats, const int* __restrict__ sel,
                         _Float16* __restrict__ X16) {
  const int slot = blockIdx.x;
  const int p    = sel[slot];            // scalar broadcast
  const int tid  = threadIdx.x;          // 128 threads
  #pragma unroll
  for (int i = 0; i < 4; ++i) {
    int ch = tid + i * 128;
    _Float16 v = (_Float16)0.0f;
    if (p >= 0) v = (_Float16)feats[(size_t)ch * HW + p];
    X16[(size_t)slot * CCH + ch] = v;
  }
}

// ---------------------------------------------------------------------------
// 5) WMMA GEMM:  C[m][n] = act( sum_k A[m][k]*W[n][k] + bias[n] )
//    A: MxK f16 row-major.  W: NxK f16 row-major (B[k][n] = W[n][k]).
//    Block = 128 threads (4 waves). Block tile 64x64, wave tile 32x32
//    (2 A-frags x 2 B-frags -> 4 v_wmma per 32-k step). Double-buffered LDS:
//    global loads for step k+1 issue before the WMMAs of step k, one barrier
//    per step. LDS pair-rows padded to 68 dwords -> conflict-free ds access.
// ---------------------------------------------------------------------------
template <bool LEAKY, bool OUT16>
__global__ void k_gemm(const _Float16* __restrict__ A, const _Float16* __restrict__ W,
                       const float* __restrict__ bias,
                       _Float16* __restrict__ C16, float* __restrict__ C32,
                       int K, int N) {
  const int tid  = threadIdx.x;       // 128 threads
  const int lane = tid & 31;
  const int wave = tid >> 5;          // 0..3
  const int lg   = lane >> 4;         // lane group 0/1
  const int lm   = lane & 15;
  const int m0   = blockIdx.x * 64;
  const int n0b  = blockIdx.y * 64;
  const int mh   = (wave & 1) * 32;   // wave m-offset in block tile
  const int nh   = (wave >> 1) * 32;  // wave n-offset in block tile

  constexpr int LDP = 68;             // padded pair-row stride (dwords)
  __shared__ unsigned int As[2][16 * LDP];   // [kpair][m], k stepped by 2
  __shared__ unsigned int Bs[2][16 * LDP];   // [kpair][n]

  v8f acc00 = {}, acc01 = {}, acc10 = {}, acc11 = {};

  uint4 ra[2], rb[2];
  const int srow = tid >> 2;          // 0..31 (+32 for t=1)
  const int squad = tid & 3;

  auto ldglobal = [&](int k0) {
    #pragma unroll
    for (int t = 0; t < 2; ++t) {
      int r = srow + t * 32;
      ra[t] = *(const uint4*)(A + (size_t)(m0 + r) * K + k0 + 8 * squad);
      rb[t] = *(const uint4*)(W + (size_t)(n0b + r) * K + k0 + 8 * squad);
      __builtin_prefetch((const void*)(A + (size_t)(m0 + r) * K + k0 + 32), 0, 1);
      __builtin_prefetch((const void*)(W + (size_t)(n0b + r) * K + k0 + 32), 0, 1);
    }
  };
  auto stlds = [&](int buf) {
    #pragma unroll
    for (int t = 0; t < 2; ++t) {
      int r = srow + t * 32;
      As[buf][(4 * squad + 0) * LDP + r] = ra[t].x;
      As[buf][(4 * squad + 1) * LDP + r] = ra[t].y;
      As[buf][(4 * squad + 2) * LDP + r] = ra[t].z;
      As[buf][(4 * squad + 3) * LDP + r] = ra[t].w;
      Bs[buf][(4 * squad + 0) * LDP + r] = rb[t].x;
      Bs[buf][(4 * squad + 1) * LDP + r] = rb[t].y;
      Bs[buf][(4 * squad + 2) * LDP + r] = rb[t].z;
      Bs[buf][(4 * squad + 3) * LDP + r] = rb[t].w;
    }
  };

  // prologue: stage k-step 0
  ldglobal(0);
  stlds(0);
  __syncthreads();

  int cur = 0;
  for (int k0 = 0; k0 < K; k0 += 32) {
    const bool more = (k0 + 32) < K;
    if (more) ldglobal(k0 + 32);      // overlap next tile's loads with WMMA

    // fragment assembly per ISA 7.12.2 16-bit layouts
    FragU a0, a1, b0, b1;
    #pragma unroll
    for (int v = 0; v < 8; ++v) {
      int kpA = (v < 4) ? (v + 4 * lg) : (8 + (v - 4) + 4 * lg);  // A 16x32 table
      a0.u[v] = As[cur][kpA * LDP + mh + lm];
      a1.u[v] = As[cur][kpA * LDP + mh + 16 + lm];
      int kpB = kpA;                                              // B 32x16 table
      b0.u[v] = Bs[cur][kpB * LDP + nh + lm];
      b1.u[v] = Bs[cur][kpB * LDP + nh + 16 + lm];
    }
    acc00 = wmma_f16(a0, b0, acc00);
    acc01 = wmma_f16(a0, b1, acc01);
    acc10 = wmma_f16(a1, b0, acc10);
    acc11 = wmma_f16(a1, b1, acc11);

    if (more) { stlds(cur ^ 1); __syncthreads(); cur ^= 1; }
  }

  // epilogue: bias + activation + store (C/D layout: m = r + 8*lg, n = lm)
  const float bv0 = bias[n0b + nh + lm];
  const float bv1 = bias[n0b + nh + 16 + lm];
  auto epi = [&](const v8f& a, int mo, int no, float bv) {
    #pragma unroll
    for (int r = 0; r < 8; ++r) {
      float v = a[r] + bv;
      if (LEAKY) v = (v >= 0.0f) ? v : 0.2f * v;
      int m = m0 + mo + r + 8 * lg;
      int n = n0b + no + lm;
      if (OUT16) C16[(size_t)m * N + n] = (_Float16)v;
      else       C32[(size_t)m * N + n] = v;
    }
  };
  epi(acc00, mh,      nh,      bv0);
  epi(acc01, mh,      nh + 16, bv1);
  epi(acc10, mh + 16, nh,      bv0);
  epi(acc11, mh + 16, nh + 16, bv1);
}

// ---------------------------------------------------------------------------
// 6) L2 row-normalize: out = h / max(||h||, 1e-12)
// ---------------------------------------------------------------------------
__global__ void k_norm(const float* __restrict__ H3, float* __restrict__ out) {
  const int row = blockIdx.x;
  const int tid = threadIdx.x;                 // 256 threads, one per column
  __shared__ float red[256];
  float v = H3[(size_t)row * DOUT + tid];
  red[tid] = v * v;
  __syncthreads();
  for (int s = 128; s > 0; s >>= 1) { if (tid < s) red[tid] += red[tid + s]; __syncthreads(); }
  float nrm = fmaxf(sqrtf(red[0]), 1e-12f);
  out[(size_t)row * DOUT + tid] = v / nrm;
}

// ---------------------------------------------------------------------------
extern "C" void kernel_launch(void* const* d_in, const int* in_sizes, int n_in,
                              void* d_out, int out_size, void* d_ws, size_t ws_size,
                              hipStream_t stream) {
  const float* feats   = (const float*)d_in[0];
  const float* labels  = (const float*)d_in[1];
  const int*   pred    = (const int*)  d_in[2];
  const float* w1      = (const float*)d_in[3];
  const float* b1      = (const float*)d_in[4];
  const float* w2      = (const float*)d_in[5];
  const float* b2      = (const float*)d_in[6];
  const float* w3      = (const float*)d_in[7];
  const float* b3      = (const float*)d_in[8];

  float* out_anchor = (float*)d_out;
  int*   out_y      = (int*)((float*)d_out + (size_t)TOT * DOUT);

  char* ws = (char*)d_ws;
  size_t off = 0;
  auto carve = [&](size_t bytes) { size_t r = off; off += (bytes + 255) & ~(size_t)255; return r; };

  int*       yhat = (int*)      (ws + carve((size_t)HW * 4));
  int*       sel  = (int*)      (ws + carve((size_t)TOT * 4));
  _Float16*  X16  = (_Float16*) (ws + carve((size_t)TOT * CCH * 2));
  _Float16*  W1h  = (_Float16*) (ws + carve((size_t)512 * 512 * 2));
  _Float16*  W2h  = (_Float16*) (ws + carve((size_t)256 * 512 * 2));
  _Float16*  W3h  = (_Float16*) (ws + carve((size_t)256 * 256 * 2));
  _Float16*  H1   = (_Float16*) (ws + carve((size_t)TOT * 512 * 2));
  _Float16*  H2   = (_Float16*) (ws + carve((size_t)TOT * 256 * 2));
  float*     H3   = (float*)    (ws + carve((size_t)TOT * 256 * 4));

  k_yhat  <<<HW / 256, 256, 0, stream>>>(labels, yhat);
  k_select<<<NCLASS, 256, 0, stream>>>(yhat, pred, sel, out_y);
  k_wconv <<<(458752 + 255) / 256, 256, 0, stream>>>(w1, w2, w3, W1h, W2h, W3h);
  k_gather<<<TOT, 128, 0, stream>>>(feats, sel, X16);

  // layer 1: 12288x512 @ (512x512)^T, leaky, f16 out
  k_gemm<true,  true ><<<dim3(TOT / 64, 512 / 64), 128, 0, stream>>>(X16, W1h, b1, H1, nullptr, 512, 512);
  // layer 2: 12288x512 @ (256x512)^T, leaky, f16 out
  k_gemm<true,  true ><<<dim3(TOT / 64, 256 / 64), 128, 0, stream>>>(H1,  W2h, b2, H2, nullptr, 512, 256);
  // layer 3: 12288x256 @ (256x256)^T, linear, f32 out
  k_gemm<false, false><<<dim3(TOT / 64, 256 / 64), 128, 0, stream>>>(H2,  W3h, b3, nullptr, H3, 256, 256);

  k_norm<<<TOT, DOUT, 0, stream>>>(H3, out_anchor);
}
